// glassDBN_75453985456601
// MI455X (gfx1250) — compile-verified, matrix operand order checked by reading, not run on hardware
//
#include <hip/hip_runtime.h>
#include <hip/hip_bf16.h>
#include <math.h>

typedef __attribute__((ext_vector_type(8)))  _Float16 v8h;
typedef __attribute__((ext_vector_type(16))) _Float16 v16h;
typedef __attribute__((ext_vector_type(8)))  float    v8f;

#define Bn   64
#define Tn   128
#define Cn   53
#define En   64
#define Hn   64
#define OUTn 2
#define ROWS (Bn * Cn)        /* 3392 = 53 * 64 */
#define C2n  (Cn * Cn)        /* 2809 */
#define H1n  1404
#define H2n  702
#define C2P  2816             /* 2809 -> mult of 64 */
#define H1P  1408
#define H2P  704

// ---------------------------------------------------------------------------
// WMMA GEMM, 4 waves / 128-thread block, each wave one 16x16 f32 tile of a
// 64x16 macro-tile:  C = A(MxKpad f16, lda) * W^T + bias, optional ReLU.
// W is N x Kpad row-major (ldb).  K-padding of A and W MUST be zero.
// Reads are unguarded (rows/cols beyond M/N only feed unstored outputs);
// stores are guarded by M/N.  Kpad must be a multiple of 64; lda/ldb mult of 8.
// blockIdx.z = batch with element strides sA/sB/sC32/sC16.
// ---------------------------------------------------------------------------
__device__ __forceinline__ v16h frag16(const _Float16* __restrict__ row, int kb, int hf) {
    // lane layout (16-bit A/B 16x32): elems 0..7  = k + {0..7} + 8*half
    //                                 elems 8..15 = k + 16 + {0..7} + 8*half
    v8h lo = *(const v8h*)(row + kb + 8 * hf);
    v8h hi = *(const v8h*)(row + kb + 16 + 8 * hf);
    return __builtin_shufflevector(lo, hi, 0, 1, 2, 3, 4, 5, 6, 7,
                                   8, 9, 10, 11, 12, 13, 14, 15);
}

__global__ __launch_bounds__(128)
void k_wmma_gemm(const _Float16* __restrict__ A, int lda, long sA,
                 const _Float16* __restrict__ W, int ldb, long sB,
                 const float* __restrict__ bias,
                 float* __restrict__ C32, int ldc, long sC32,
                 _Float16* __restrict__ C16, int ldc16, long sC16,
                 int M, int N, int Kpad, int relu) {
    const int lane = threadIdx.x & 31;
    const int wv   = threadIdx.x >> 5;
    const int hf   = lane >> 4;
    const int l15  = lane & 15;
    const int m0   = (blockIdx.x * 4 + wv) * 16;
    const int n0   = blockIdx.y * 16;
    const long z   = blockIdx.z;
    const _Float16* rowA = A + z * sA + (long)(m0 + l15) * lda;
    const _Float16* rowW = W + z * sB + (long)(n0 + l15) * ldb;

    v8f acc = {0.f, 0.f, 0.f, 0.f, 0.f, 0.f, 0.f, 0.f};
    for (int kb = 0; kb < Kpad; kb += 64) {
        if (kb + 64 < Kpad) {
            __builtin_prefetch((const void*)(rowA + kb + 64), 0, 3);
            __builtin_prefetch((const void*)(rowW + kb + 64), 0, 3);
        }
        v16h a0 = frag16(rowA, kb, hf);
        v16h b0 = frag16(rowW, kb, hf);
        v16h a1 = frag16(rowA, kb + 32, hf);
        v16h b1 = frag16(rowW, kb + 32, hf);
        acc = __builtin_amdgcn_wmma_f32_16x16x32_f16(false, a0, false, b0,
                                                     (short)0, acc, false, false);
        acc = __builtin_amdgcn_wmma_f32_16x16x32_f16(false, a1, false, b1,
                                                     (short)0, acc, false, false);
    }

    const int nB = n0 + l15;
    float bval = (bias != nullptr && nB < N) ? bias[nB] : 0.f;
#pragma unroll
    for (int r = 0; r < 8; ++r) {
        const int m = m0 + r + 8 * hf;       // C/D layout: VGPR r -> M = r + 8*half
        float vv = acc[r] + bval;
        if (relu) vv = fmaxf(vv, 0.f);
        if (m < M && nB < N) {
            if (C32) C32[z * sC32 + (long)m * ldc + nB]   = vv;
            if (C16) C16[z * sC16 + (long)m * ldc16 + nB] = (_Float16)vv;
        }
    }
}

// ---------------------------------------------------------------------------
// Elementwise / small kernels
// ---------------------------------------------------------------------------
__global__ void k_zero_h(float* h32, _Float16* h16, int n) {
    int i = blockIdx.x * blockDim.x + threadIdx.x;
    if (i < n) { h32[i] = 0.f; h16[i] = (_Float16)0.f; }
}

__global__ void k_zero_f16(_Float16* d, int n) {
    int i = blockIdx.x * blockDim.x + threadIdx.x;
    if (i < n) d[i] = (_Float16)0.f;
}

__global__ void k_f32_to_f16(const float* __restrict__ s, _Float16* __restrict__ d, int n) {
    int i = blockIdx.x * blockDim.x + threadIdx.x;
    if (i < n) d[i] = (_Float16)s[i];
}

// Pad-convert weight: dst is Npad x ld (f16); valid region N x K from src (NxK f32).
__global__ void k_pad_f16(const float* __restrict__ s, _Float16* __restrict__ d,
                          int N, int Npad, int K, int ld) {
    int i = blockIdx.x * blockDim.x + threadIdx.x;
    if (i >= Npad * ld) return;
    int n = i / ld, k = i - n * ld;
    d[i] = (n < N && k < K) ? (_Float16)s[(long)n * K + k] : (_Float16)0.f;
}

__global__ void k_pad_bias(const float* __restrict__ s, float* __restrict__ d,
                           int N, int Npad) {
    int i = blockIdx.x * blockDim.x + threadIdx.x;
    if (i < Npad) d[i] = (i < N) ? s[i] : 0.f;
}

__global__ void k_embed(const float* __restrict__ x, const float* __restrict__ ew,
                        const float* __restrict__ eb, _Float16* __restrict__ a, int t) {
    int i = blockIdx.x * blockDim.x + threadIdx.x;
    if (i >= ROWS * En) return;
    int r = i >> 6, k = i & 63;
    int b = r / Cn, c = r - b * Cn;
    float xv = x[((long)b * Tn + t) * Cn + c];
    a[i] = (_Float16)(xv * ew[k] + eb[k]);
}

__device__ __forceinline__ float sigf(float x) { return 1.f / (1.f + expf(-x)); }

// GRU gate fusion; also emits per-batch transposed f16 copy hnT[b][j][c]
// (64x64 tiles; c >= 53 columns stay zero from the one-time init).
__global__ void k_gru_combine(const float* __restrict__ gi, const float* __restrict__ gh,
                              const float* __restrict__ hOld,
                              float* __restrict__ hn32, _Float16* __restrict__ hn16,
                              _Float16* __restrict__ hnT16) {
    int i = blockIdx.x * blockDim.x + threadIdx.x;
    if (i >= ROWS * Hn) return;
    int r = i >> 6, j = i & 63;
    int b = r / Cn, c = r - b * Cn;
    const float* giR = gi + (long)r * 192;
    const float* ghR = gh + (long)r * 192;
    float ir = giR[j], iz = giR[64 + j], inn = giR[128 + j];
    float hr = ghR[j], hz = ghR[64 + j], hnn = ghR[128 + j];
    float rr = sigf(ir + hr);
    float zz = sigf(iz + hz);
    float nn = tanhf(inn + rr * hnn);
    float hv = (1.f - zz) * nn + zz * hOld[i];
    hn32[i] = hv;
    hn16[i] = (_Float16)hv;
    hnT16[(long)b * 4096 + j * 64 + c] = (_Float16)hv;
}

// Per-batch Frobenius-normalize + gate; emit f16 padded transfer (64x64) and the
// f16 classifier row (2816-wide, tail zeroed so clf GEMM can run unguarded K).
__global__ void k_norm_gate(float* __restrict__ mixBase, const float* __restrict__ gateB,
                            _Float16* __restrict__ trans16, _Float16* __restrict__ clfA,
                            int t) {
    const int b   = blockIdx.x;
    const int tid = threadIdx.x;
    float* mix = mixBase + ((long)b * Tn + t) * C2n;
    __shared__ float red[256];
    float ss = 0.f;
    for (int i = tid; i < C2n; i += 256) { float v = mix[i]; ss += v * v; }
    red[tid] = ss;
    __syncthreads();
    for (int s = 128; s > 0; s >>= 1) {
        if (tid < s) red[tid] += red[tid + s];
        __syncthreads();
    }
    const float inv = 1.f / sqrtf(red[0]);
    for (int idx = tid; idx < 64 * 64; idx += 256) {
        int c = idx >> 6, d = idx & 63;
        float out = 0.f;
        if (c < Cn && d < Cn) {
            int e = c * Cn + d;
            float v = mix[e] * inv;
            float g = sigf(fabsf(v) + gateB[e]);
            out = v * g;
            mix[e] = out;
            clfA[(long)b * C2P + e] = (_Float16)out;
        }
        trans16[(long)b * 4096 + idx] = (_Float16)out;
    }
    if (tid < C2P - C2n) clfA[(long)b * C2P + C2n + tid] = (_Float16)0.f;
}

__global__ void k_pred(const float* __restrict__ h, const float* __restrict__ pw,
                       const float* __restrict__ pb, float* __restrict__ predOut, int t) {
    int r = blockIdx.x * blockDim.x + threadIdx.x;
    if (r >= ROWS) return;
    const float* hr = h + (long)r * Hn;
    float s = pb[0];
    for (int j = 0; j < Hn; ++j) s += hr[j] * pw[j];
    int b = r / Cn, c = r - b * Cn;
    predOut[((long)b * (Tn - 1) + t) * Cn + c] = s;
}

__global__ void k_clf3(const float* __restrict__ h2, const float* __restrict__ w3,
                       const float* __restrict__ b3, float* __restrict__ tl, int t) {
    const int b = blockIdx.x, tid = threadIdx.x;
    __shared__ float a0[128], a1[128];
    float s0 = 0.f, s1 = 0.f;
    const float* row = h2 + (long)b * H2P;
    for (int k = tid; k < H2n; k += 128) {
        float v = row[k];
        s0 += v * w3[k];
        s1 += v * w3[H2n + k];
    }
    a0[tid] = s0; a1[tid] = s1;
    __syncthreads();
    for (int s = 64; s > 0; s >>= 1) {
        if (tid < s) { a0[tid] += a0[tid + s]; a1[tid] += a1[tid + s]; }
        __syncthreads();
    }
    if (tid == 0) {
        long off = ((long)b * Tn + t) * OUTn;
        tl[off + 0] = a0[0] + b3[0];
        tl[off + 1] = a1[0] + b3[1];
    }
}

__global__ void k_mean_logits(const float* __restrict__ tl, float* __restrict__ logits) {
    int i = blockIdx.x * blockDim.x + threadIdx.x;
    if (i >= Bn * OUTn) return;
    int b = i >> 1, o = i & 1;
    float s = 0.f;
    for (int t = 0; t < Tn; ++t) s += tl[((long)b * Tn + t) * OUTn + o];
    logits[i] = s * (1.f / (float)Tn);
}

// ---------------------------------------------------------------------------
// Launch
// ---------------------------------------------------------------------------
extern "C" void kernel_launch(void* const* d_in, const int* in_sizes, int n_in,
                              void* d_out, int out_size, void* d_ws, size_t ws_size,
                              hipStream_t stream) {
    (void)in_sizes; (void)n_in; (void)out_size; (void)ws_size;
    const float* x        = (const float*)d_in[0];
    const float* emb_w    = (const float*)d_in[1];
    const float* emb_b    = (const float*)d_in[2];
    const float* gru_wih  = (const float*)d_in[3];
    const float* gru_whh  = (const float*)d_in[4];
    const float* gru_bih  = (const float*)d_in[5];
    const float* gru_bhh  = (const float*)d_in[6];
    const float* q_w1 = (const float*)d_in[7];  const float* q_b1 = (const float*)d_in[8];
    const float* q_w2 = (const float*)d_in[9];  const float* q_b2 = (const float*)d_in[10];
    const float* q_w3 = (const float*)d_in[11]; const float* q_b3 = (const float*)d_in[12];
    const float* k_w1 = (const float*)d_in[13]; const float* k_b1 = (const float*)d_in[14];
    const float* k_w2 = (const float*)d_in[15]; const float* k_b2 = (const float*)d_in[16];
    const float* k_w3 = (const float*)d_in[17]; const float* k_b3 = (const float*)d_in[18];
    const float* gate_bias= (const float*)d_in[19];
    const float* pred_w   = (const float*)d_in[20];
    const float* pred_b   = (const float*)d_in[21];
    const float* clf_w1   = (const float*)d_in[22]; const float* clf_b1 = (const float*)d_in[23];
    const float* clf_w2   = (const float*)d_in[24]; const float* clf_b2 = (const float*)d_in[25];
    const float* clf_w3   = (const float*)d_in[26]; const float* clf_b3 = (const float*)d_in[27];

    float* outF = (float*)d_out;
    const long mixOff  = 128;
    const long tlOff   = mixOff + (long)Bn * Tn * C2n;        // 23,011,456
    const long predOff = tlOff + (long)Bn * Tn * OUTn;        // 23,027,840
    float* mixBase  = outF + mixOff;
    float* tlBase   = outF + tlOff;
    float* predBase = outF + predOff;

    // ---- bump allocator on d_ws (everything 256B aligned) ----
    size_t off = 0;
    auto alloc = [&](size_t bytes) -> void* {
        void* p = (char*)d_ws + off;
        off += (bytes + 255) & ~(size_t)255;
        return p;
    };
    const size_t PAD = 1024;  // 16 rows x 64 over-read pad for per-batch tiles
    _Float16* wih16   = (_Float16*)alloc(192 * 64 * sizeof(_Float16));
    _Float16* whh16   = (_Float16*)alloc(192 * 64 * sizeof(_Float16));
    _Float16* qw1h    = (_Float16*)alloc(4096 * sizeof(_Float16));
    _Float16* qw2h    = (_Float16*)alloc(4096 * sizeof(_Float16));
    _Float16* qw3h    = (_Float16*)alloc(4096 * sizeof(_Float16));
    _Float16* kw1h    = (_Float16*)alloc(4096 * sizeof(_Float16));
    _Float16* kw2h    = (_Float16*)alloc(4096 * sizeof(_Float16));
    _Float16* kw3h    = (_Float16*)alloc(4096 * sizeof(_Float16));
    _Float16* clfW1p  = (_Float16*)alloc((size_t)H1P * C2P * sizeof(_Float16));
    _Float16* clfW2p  = (_Float16*)alloc((size_t)H2P * H1P * sizeof(_Float16));
    _Float16* embA    = (_Float16*)alloc((size_t)ROWS * En * sizeof(_Float16));
    _Float16* hF16    = (_Float16*)alloc((size_t)ROWS * Hn * sizeof(_Float16));
    _Float16* hnF16   = (_Float16*)alloc((size_t)ROWS * Hn * sizeof(_Float16));
    _Float16* hnT16   = (_Float16*)alloc((size_t)Bn * 4096 * sizeof(_Float16));
    _Float16* tA16    = (_Float16*)alloc((size_t)ROWS * Hn * sizeof(_Float16));
    _Float16* tB16    = (_Float16*)alloc((size_t)ROWS * Hn * sizeof(_Float16));
    _Float16* q16     = (_Float16*)alloc(((size_t)ROWS * Hn + PAD) * sizeof(_Float16));
    _Float16* k16     = (_Float16*)alloc(((size_t)ROWS * Hn + PAD) * sizeof(_Float16));
    _Float16* trans16 = (_Float16*)alloc((size_t)Bn * 4096 * sizeof(_Float16));
    _Float16* clfA    = (_Float16*)alloc((size_t)Bn * C2P * sizeof(_Float16));
    _Float16* clfH1h  = (_Float16*)alloc((size_t)Bn * H1P * sizeof(_Float16));
    float*    clfB1p  = (float*)alloc((size_t)H1P * sizeof(float));
    float*    clfB2p  = (float*)alloc((size_t)H2P * sizeof(float));
    float*    hF32    = (float*)alloc((size_t)ROWS * Hn * sizeof(float));
    float*    hnF32   = (float*)alloc((size_t)ROWS * Hn * sizeof(float));
    float*    giF32   = (float*)alloc((size_t)ROWS * 192 * sizeof(float));
    float*    ghF32   = (float*)alloc((size_t)ROWS * 192 * sizeof(float));
    float*    clfH2f  = (float*)alloc((size_t)Bn * H2P * sizeof(float));

    // ---- per-launch weight conversions / zero-inits (deterministic) ----
    auto cvt = [&](const float* s, _Float16* d, int n) {
        k_f32_to_f16<<<(n + 255) / 256, 256, 0, stream>>>(s, d, n);
    };
    cvt(gru_wih, wih16, 192 * 64);
    cvt(gru_whh, whh16, 192 * 64);
    cvt(q_w1, qw1h, 4096); cvt(q_w2, qw2h, 4096); cvt(q_w3, qw3h, 4096);
    cvt(k_w1, kw1h, 4096); cvt(k_w2, kw2h, 4096); cvt(k_w3, kw3h, 4096);
    k_pad_f16<<<((H1P * C2P) + 255) / 256, 256, 0, stream>>>(clf_w1, clfW1p, H1n, H1P, C2n, C2P);
    k_pad_f16<<<((H2P * H1P) + 255) / 256, 256, 0, stream>>>(clf_w2, clfW2p, H2n, H2P, H1n, H1P);
    k_pad_bias<<<(H1P + 255) / 256, 256, 0, stream>>>(clf_b1, clfB1p, H1n, H1P);
    k_pad_bias<<<(H2P + 255) / 256, 256, 0, stream>>>(clf_b2, clfB2p, H2n, H2P);
    k_zero_h<<<((ROWS * Hn) + 255) / 256, 256, 0, stream>>>(hF32, hF16, ROWS * Hn);
    k_zero_f16<<<((Bn * 4096) + 255) / 256, 256, 0, stream>>>(hnT16, Bn * 4096);

    const long rowStride = (long)Cn * Hn;   // per-batch stride in h/q/k buffers
    const int  EW = 256;
    const int  MG = ROWS / 64;              // 53 macro-tiles along M for row-space GEMMs

    for (int t = 0; t < Tn; ++t) {
        // embedding -> f16 A matrix (ROWS x 64)
        k_embed<<<((ROWS * En) + EW - 1) / EW, EW, 0, stream>>>(x, emb_w, emb_b, embA, t);

        // gi = embA @ Wih^T + bih     (ROWS x 192)
        k_wmma_gemm<<<dim3(MG, 12, 1), 128, 0, stream>>>(
            embA, 64, 0, wih16, 64, 0, gru_bih,
            giF32, 192, 0, nullptr, 0, 0, ROWS, 192, 64, 0);
        // gh = h @ Whh^T + bhh
        k_wmma_gemm<<<dim3(MG, 12, 1), 128, 0, stream>>>(
            hF16, 64, 0, whh16, 64, 0, gru_bhh,
            ghF32, 192, 0, nullptr, 0, 0, ROWS, 192, 64, 0);
        // GRU gates -> hnew (+ transposed per-batch f16 copy)
        k_gru_combine<<<((ROWS * Hn) + EW - 1) / EW, EW, 0, stream>>>(
            giF32, ghF32, hF32, hnF32, hnF16, hnT16);

        // q = MLP3(hnew)
        k_wmma_gemm<<<dim3(MG, 4, 1), 128, 0, stream>>>(
            hnF16, 64, 0, qw1h, 64, 0, q_b1,
            nullptr, 0, 0, tA16, 64, 0, ROWS, 64, 64, 1);
        k_wmma_gemm<<<dim3(MG, 4, 1), 128, 0, stream>>>(
            tA16, 64, 0, qw2h, 64, 0, q_b2,
            nullptr, 0, 0, tB16, 64, 0, ROWS, 64, 64, 1);
        k_wmma_gemm<<<dim3(MG, 4, 1), 128, 0, stream>>>(
            tB16, 64, 0, qw3h, 64, 0, q_b3,
            nullptr, 0, 0, q16, 64, 0, ROWS, 64, 64, 0);
        // k = MLP3(hnew)
        k_wmma_gemm<<<dim3(MG, 4, 1), 128, 0, stream>>>(
            hnF16, 64, 0, kw1h, 64, 0, k_b1,
            nullptr, 0, 0, tA16, 64, 0, ROWS, 64, 64, 1);
        k_wmma_gemm<<<dim3(MG, 4, 1), 128, 0, stream>>>(
            tA16, 64, 0, kw2h, 64, 0, k_b2,
            nullptr, 0, 0, tB16, 64, 0, ROWS, 64, 64, 1);
        k_wmma_gemm<<<dim3(MG, 4, 1), 128, 0, stream>>>(
            tB16, 64, 0, kw3h, 64, 0, k_b3,
            nullptr, 0, 0, k16, 64, 0, ROWS, 64, 64, 0);

        // transfer[b] = q_b @ k_b^T  -> raw into mixing slab of d_out
        k_wmma_gemm<<<dim3(1, 4, Bn), 128, 0, stream>>>(
            q16, 64, rowStride, k16, 64, rowStride, nullptr,
            mixBase + (long)t * C2n, Cn, (long)Tn * C2n,
            nullptr, 0, 0, Cn, Cn, 64, 0);

        // normalize + gate; emit f16 padded transfer + classifier row
        k_norm_gate<<<Bn, 256, 0, stream>>>(mixBase, gate_bias, trans16, clfA, t);

        // hmix[b] = transfer_b (53x53 in 64x64, zero-padded) @ hnew_b
        //   via W = hnT (N=h rows x K=d cols, zero-padded cols) -> new carry h
        k_wmma_gemm<<<dim3(1, 4, Bn), 128, 0, stream>>>(
            trans16, 64, 4096, hnT16, 64, 4096, nullptr,
            hF32, 64, rowStride, hF16, 64, rowStride, Cn, Hn, 64, 0);

        // predictor uses hidden[:, :-1]
        if (t < Tn - 1)
            k_pred<<<(ROWS + EW - 1) / EW, EW, 0, stream>>>(hF32, pred_w, pred_b, predBase, t);

        // classifier: (64 x 2809) -> 1404 -> 702 -> 2, fully padded pipeline
        k_wmma_gemm<<<dim3(1, H1P / 16, 1), 128, 0, stream>>>(
            clfA, C2P, 0, clfW1p, C2P, 0, clfB1p,
            nullptr, 0, 0, clfH1h, H1P, 0, Bn, H1P, C2P, 1);
        k_wmma_gemm<<<dim3(1, H2P / 16, 1), 128, 0, stream>>>(
            clfH1h, H1P, 0, clfW2p, H1P, 0, clfB2p,
            clfH2f, H2P, 0, nullptr, 0, 0, Bn, H2P, H1P, 1);
        k_clf3<<<Bn, 128, 0, stream>>>(clfH2f, clf_w3, clf_b3, tlBase, t);
    }

    // logits = mean over t of time_logits
    k_mean_logits<<<1, 128, 0, stream>>>(tlBase, outF);
}
